// TGCN_14963666059365
// MI455X (gfx1250) — compile-verified, matrix operand order checked by reading, not run on hardware
//
#include <hip/hip_runtime.h>
#include <hip/hip_bf16.h>
#include <stdint.h>

#define N_PAPER   100000
#define N_AUTHOR  50000
#define NEDGE     500000
#define FEAT      128
#define HDIM      128
#define HHALF     64
#define NCLS      8
#define NEG_SLOPE 0.01f

typedef __attribute__((ext_vector_type(16))) __bf16    bf16x16;
typedef __attribute__((ext_vector_type(8)))  float     f32x8;
typedef __attribute__((ext_vector_type(4)))  unsigned  u32x4;
typedef __attribute__((ext_vector_type(8)))  unsigned  u32x8;

static __device__ __forceinline__ float leaky(float v) {
    return v > 0.0f ? v : v * NEG_SLOPE;
}

// ---------------------------------------------------------------- TDM helper
// 1-D Tensor Data Mover copy: nwords8 * 8 bytes from global -> LDS.
// D# group0: count=1 | lds_addr | global_addr[56:0] | type=2
// D# group1: data_size=3 (8B), tensor_dim0 = tile_dim0 = dim0_stride = nwords8.
// Issued once per calling wave; completion via s_wait_tensorcnt.
static __device__ __forceinline__ void tdm_load_1d(const void* gptr, void* lptr,
                                                   unsigned nwords8) {
    unsigned long long ga = (unsigned long long)gptr;
    unsigned lo = (unsigned)(unsigned long long)lptr;   // LDS byte offset (flat addr[31:0])
    u32x4 g0;
    g0[0] = 1u;                                          // count=1 (valid user D#)
    g0[1] = lo;                                          // lds_addr
    g0[2] = (unsigned)ga;                                // global_addr[31:0]
    g0[3] = (unsigned)((ga >> 32) & 0x1FFFFFFull) | (2u << 30);  // addr[56:32] | type=2
    u32x8 g1;
    g1[0] = 0x00030000u;                                 // data_size=3 (8 bytes)
    g1[1] = (nwords8 & 0xFFFFu) << 16;                   // tensor_dim0[15:0]
    g1[2] = (nwords8 >> 16) & 0xFFFFu;                   // tensor_dim0[31:16]
    g1[3] = (nwords8 & 0xFFFFu) << 16;                   // tile_dim0 (16-bit)
    g1[4] = 0u;                                          // tile_dim1/2 unused -> 1D
    g1[5] = nwords8;                                     // tensor_dim0_stride[31:0]
    g1[6] = 0u;
    g1[7] = 0u;
    asm volatile("tensor_load_to_lds %0, %1" :: "s"(g0), "s"(g1) : "memory");
}

// ---------------------------------------------------------------- utilities
__global__ void k_zero(float4* __restrict__ p, long n4) {
    long i = (long)blockIdx.x * blockDim.x + threadIdx.x;
    long stride = (long)gridDim.x * blockDim.x;
    float4 z = make_float4(0.f, 0.f, 0.f, 0.f);
    for (; i < n4; i += stride) p[i] = z;
}

// Wt_rel[r][o][k] = sum_b coeff[r][b] * basis[b][k][o]   (N-major, bf16)
__global__ void k_build_rel_w(const float* __restrict__ basis,
                              const float* __restrict__ coeff,
                              __bf16* __restrict__ Wt) {
    int idx = blockIdx.x * blockDim.x + threadIdx.x;
    if (idx >= 3 * HDIM * FEAT) return;
    int r   = idx / (HDIM * FEAT);
    int rem = idx - r * HDIM * FEAT;
    int o   = rem >> 7;
    int k   = rem & 127;
    float w = coeff[r * 2 + 0] * basis[(0 * FEAT + k) * HDIM + o]
            + coeff[r * 2 + 1] * basis[(1 * FEAT + k) * HDIM + o];
    Wt[idx] = (__bf16)w;
}

// Wt[n][k] = (bf16) W[k][n]    (W is [K,N] row-major fp32)
__global__ void k_transpose_w(const float* __restrict__ W, __bf16* __restrict__ Wt,
                              int K, int N) {
    int idx = blockIdx.x * blockDim.x + threadIdx.x;
    if (idx >= K * N) return;
    int n = idx / K, k = idx - n * K;
    Wt[idx] = (__bf16)W[k * N + n];
}

// ------------------------------------------------- WMMA fragment loaders
// 16-bit A 16x32: lanes 0-15 -> rows M, elems 0..7 = K(half*8+i), 8..15 = K(16+half*8+i).
// B mirrors with N = lane&15 against the N-major (pre-transposed) weight panel.
static __device__ __forceinline__ bf16x16 load_a_f32(const float* ap) {
    bf16x16 a;
#pragma unroll
    for (int i = 0; i < 8; ++i) a[i] = (__bf16)ap[i];
#pragma unroll
    for (int i = 0; i < 8; ++i) a[8 + i] = (__bf16)ap[16 + i];
    return a;
}
static __device__ __forceinline__ bf16x16 load_a_f32_scaled(const float* ap, float s) {
    bf16x16 a;
#pragma unroll
    for (int i = 0; i < 8; ++i) a[i] = (__bf16)(ap[i] * s);
#pragma unroll
    for (int i = 0; i < 8; ++i) a[8 + i] = (__bf16)(ap[16 + i] * s);
    return a;
}
static __device__ __forceinline__ bf16x16 load_b_lds(const __bf16* bp) {
    bf16x16 b;
#pragma unroll
    for (int i = 0; i < 8; ++i) b[i] = bp[i];
#pragma unroll
    for (int i = 0; i < 8; ++i) b[8 + i] = bp[16 + i];
    return b;
}

// --------------------------------------------- input projection (X @ Wt^T + b)
// Weight panel staged into LDS by the Tensor Data Mover; one wave per 16-row
// strip, 8 column tiles. B fragments preloaded per K-step, WMMAs back-to-back.
__global__ void k_proj_gemm(const float* __restrict__ X, const __bf16* __restrict__ Wt,
                            const float* __restrict__ bias, __bf16* __restrict__ Y,
                            int M, int K) {
    __shared__ __bf16 sWt[HDIM * FEAT];
    if ((threadIdx.x >> 5) == 0) {                     // wave 0 drives the TDM
        tdm_load_1d(Wt, sWt, (unsigned)(HDIM * K * 2 / 8));
        __builtin_amdgcn_s_wait_tensorcnt(0);
    }
    __syncthreads();

    int wave  = threadIdx.x >> 5;
    int strip = blockIdx.x * (blockDim.x >> 5) + wave;
    if (strip >= (M >> 4)) return;                     // uniform per-wave exit (post-barrier)
    int lane = threadIdx.x & 31;
    int half = lane >> 4;
    int mr   = lane & 15;
    int row  = (strip << 4) + mr;

    f32x8 acc[8];
    f32x8 zero = {0.f, 0.f, 0.f, 0.f, 0.f, 0.f, 0.f, 0.f};
#pragma unroll
    for (int t = 0; t < 8; ++t) acc[t] = zero;

    const float* arow = X + (long)row * K;
    for (int k0 = 0; k0 < K; k0 += 32) {
        bf16x16 a = load_a_f32(arow + k0 + half * 8);
        bf16x16 bfr[8];
#pragma unroll
        for (int t = 0; t < 8; ++t)
            bfr[t] = load_b_lds(sWt + (t * 16 + mr) * K + k0 + half * 8);
#pragma unroll
        for (int t = 0; t < 8; ++t)
            acc[t] = __builtin_amdgcn_wmma_f32_16x16x32_bf16(
                false, a, false, bfr[t], (short)0, acc[t], false, false);
    }
    int rbase = (strip << 4) + half * 8;               // D: vgpr v -> M = v (+8 hi half)
#pragma unroll
    for (int t = 0; t < 8; ++t) {
        int col  = t * 16 + mr;
        float bv = bias[col];
#pragma unroll
        for (int v = 0; v < 8; ++v)
            Y[(long)(rbase + v) * HDIM + col] = (__bf16)(acc[t][v] + bv);
    }
}

// --------------------------------------------- edge scatter (segment sum)
__global__ void k_scatter(const __bf16* __restrict__ Hsrc,
                          const int* __restrict__ src, const int* __restrict__ dst,
                          float* __restrict__ agg, float* __restrict__ deg, int nE) {
    unsigned tid = blockIdx.x * blockDim.x + threadIdx.x;
    int e = (int)(tid >> 5);
    if (e >= nE) return;
    int lane = threadIdx.x & 31;
    int s = src[e], d = dst[e];

    int pf = e + 4096;                                 // prefetch a future source row
    if (pf < nE) {
        int ps = src[pf];
        __builtin_prefetch(Hsrc + (long)ps * HDIM + lane * 4, 0, 1);
    }

    const __bf16* hp = Hsrc + (long)s * HDIM + lane * 4;
    float* ap = agg + (long)d * HDIM + lane * 4;
    float v0 = (float)hp[0], v1 = (float)hp[1], v2 = (float)hp[2], v3 = (float)hp[3];
    atomicAdd(ap + 0, v0);
    atomicAdd(ap + 1, v1);
    atomicAdd(ap + 2, v2);
    atomicAdd(ap + 3, v3);
    if (lane == 0) atomicAdd(deg + d, 1.0f);
}

// ------------- fused: (agg/deg) @ W_rel (+ second relation) + conv_bias, leaky
// Both relation weight panels staged to LDS via TDM. `out` may alias aggA:
// each wave reads only its own 16 rows and stores after all its loads.
__global__ void k_rel_gemm(const float* aggA, const float* __restrict__ degA,
                           const float* aggB, const float* __restrict__ degB,
                           const __bf16* __restrict__ WtA, const __bf16* __restrict__ WtB,
                           const float* __restrict__ conv_bias,
                           float* out, int M, int two) {
    __shared__ __bf16 sWt[2 * HDIM * HDIM];            // 64 KB of the 320 KB WGP LDS
    if ((threadIdx.x >> 5) == 0) {
        tdm_load_1d(WtA, sWt, (unsigned)(HDIM * HDIM * 2 / 8));
        if (two)
            tdm_load_1d(WtB, sWt + HDIM * HDIM, (unsigned)(HDIM * HDIM * 2 / 8));
        __builtin_amdgcn_s_wait_tensorcnt(0);
    }
    __syncthreads();

    int wave  = threadIdx.x >> 5;
    int strip = blockIdx.x * (blockDim.x >> 5) + wave;
    if (strip >= (M >> 4)) return;
    int lane = threadIdx.x & 31;
    int half = lane >> 4;
    int mr   = lane & 15;
    int row  = (strip << 4) + mr;

    f32x8 acc[8];
    f32x8 zero = {0.f, 0.f, 0.f, 0.f, 0.f, 0.f, 0.f, 0.f};
#pragma unroll
    for (int t = 0; t < 8; ++t) acc[t] = zero;

    int nrel = two ? 2 : 1;
    for (int rel = 0; rel < nrel; ++rel) {
        const float*  agg = rel ? aggB : aggA;
        const float*  deg = rel ? degB : degA;
        const __bf16* sW  = sWt + rel * HDIM * HDIM;
        float rinv = 1.0f / fmaxf(deg[row], 1.0f);
        const float* arow = agg + (long)row * HDIM;
        for (int k0 = 0; k0 < HDIM; k0 += 32) {
            bf16x16 a = load_a_f32_scaled(arow + k0 + half * 8, rinv);
            bf16x16 bfr[8];
#pragma unroll
            for (int t = 0; t < 8; ++t)
                bfr[t] = load_b_lds(sW + (t * 16 + mr) * HDIM + k0 + half * 8);
#pragma unroll
            for (int t = 0; t < 8; ++t)
                acc[t] = __builtin_amdgcn_wmma_f32_16x16x32_bf16(
                    false, a, false, bfr[t], (short)0, acc[t], false, false);
        }
    }
    int rbase = (strip << 4) + half * 8;
#pragma unroll
    for (int t = 0; t < 8; ++t) {
        int col  = t * 16 + mr;
        float bv = conv_bias[col];
#pragma unroll
        for (int v = 0; v < 8; ++v)
            out[(long)(rbase + v) * HDIM + col] = leaky(acc[t][v] + bv);
    }
}

// ----------------------------- MLP head: 128 -> leaky(64) -> 8 -> softmax
__global__ void k_head(const float* __restrict__ X,
                       const float* __restrict__ W1, const float* __restrict__ b1,
                       const float* __restrict__ W2, const float* __restrict__ b2,
                       float* __restrict__ Y, int M) {
    __shared__ float sW1[HDIM * HHALF];
    __shared__ float sW2[HHALF * NCLS];
    __shared__ float sb1[HHALF];
    __shared__ float sb2[NCLS];
    for (int i = threadIdx.x; i < HDIM * HHALF; i += blockDim.x) sW1[i] = W1[i];
    for (int i = threadIdx.x; i < HHALF * NCLS; i += blockDim.x) sW2[i] = W2[i];
    if (threadIdx.x < HHALF) sb1[threadIdx.x] = b1[threadIdx.x];
    if (threadIdx.x < NCLS)  sb2[threadIdx.x] = b2[threadIdx.x];
    __syncthreads();
    int r = blockIdx.x * blockDim.x + threadIdx.x;
    if (r >= M) return;

    float h1[HHALF];
#pragma unroll
    for (int j = 0; j < HHALF; ++j) h1[j] = sb1[j];
    const float* xr = X + (long)r * HDIM;
    for (int k = 0; k < HDIM; ++k) {
        float x = xr[k];
#pragma unroll
        for (int j = 0; j < HHALF; ++j) h1[j] = fmaf(x, sW1[k * HHALF + j], h1[j]);
    }
    float y[NCLS];
#pragma unroll
    for (int c = 0; c < NCLS; ++c) y[c] = sb2[c];
    for (int j = 0; j < HHALF; ++j) {
        float v = leaky(h1[j]);
#pragma unroll
        for (int c = 0; c < NCLS; ++c) y[c] = fmaf(v, sW2[j * NCLS + c], y[c]);
    }
    float mx = y[0];
#pragma unroll
    for (int c = 1; c < NCLS; ++c) mx = fmaxf(mx, y[c]);
    float ssum = 0.f;
#pragma unroll
    for (int c = 0; c < NCLS; ++c) { y[c] = __expf(y[c] - mx); ssum += y[c]; }
    float inv = 1.0f / ssum;
#pragma unroll
    for (int c = 0; c < NCLS; ++c) Y[(long)r * NCLS + c] = y[c] * inv;
}

// ------------------------------------------------------------------ driver
extern "C" void kernel_launch(void* const* d_in, const int* in_sizes, int n_in,
                              void* d_out, int out_size, void* d_ws, size_t ws_size,
                              hipStream_t stream) {
    const float* feat_paper   = (const float*)d_in[0];
    const float* feat_author  = (const float*)d_in[1];
    const int*   cites_src    = (const int*)d_in[2];
    const int*   cites_dst    = (const int*)d_in[3];
    const int*   writes_src   = (const int*)d_in[4];
    const int*   writes_dst   = (const int*)d_in[5];
    const int*   wb_src       = (const int*)d_in[6];
    const int*   wb_dst       = (const int*)d_in[7];
    const float* inp_W_paper  = (const float*)d_in[8];
    const float* inp_b_paper  = (const float*)d_in[9];
    const float* inp_W_author = (const float*)d_in[10];
    const float* inp_b_author = (const float*)d_in[11];
    const float* basis        = (const float*)d_in[12];
    const float* coeff        = (const float*)d_in[13];
    const float* conv_bias    = (const float*)d_in[14];
    const float* W1_p = (const float*)d_in[15];
    const float* b1_p = (const float*)d_in[16];
    const float* W2_p = (const float*)d_in[17];
    const float* b2_p = (const float*)d_in[18];
    const float* W1_a = (const float*)d_in[19];
    const float* b1_a = (const float*)d_in[20];
    const float* W2_a = (const float*)d_in[21];
    const float* b2_a = (const float*)d_in[22];

    char* ws = (char*)d_ws;
    size_t off = 0;
    auto take = [&](size_t bytes) -> char* {
        char* p = ws + off;
        off = (off + bytes + 255) & ~(size_t)255;
        return p;
    };

    // fp32 accumulation buffers (zeroed each call); out_p/out_a alias agg_c/agg_b
    float*  agg_c  = (float*)take((size_t)N_PAPER  * HDIM * 4);
    float*  agg_w  = (float*)take((size_t)N_PAPER  * HDIM * 4);
    float*  agg_b  = (float*)take((size_t)N_AUTHOR * HDIM * 4);
    float*  deg_c  = (float*)take((size_t)N_PAPER  * 4);
    float*  deg_w  = (float*)take((size_t)N_PAPER  * 4);
    float*  deg_b  = (float*)take((size_t)N_AUTHOR * 4);
    __bf16* hp     = (__bf16*)take((size_t)N_PAPER  * HDIM * 2);
    __bf16* ha     = (__bf16*)take((size_t)N_AUTHOR * HDIM * 2);
    __bf16* Wt_rel = (__bf16*)take((size_t)3 * HDIM * FEAT * 2);
    __bf16* Wt_ip  = (__bf16*)take((size_t)HDIM * FEAT * 2);
    __bf16* Wt_ia  = (__bf16*)take((size_t)HDIM * 64 * 2);

    // 1) zero accumulation region (agg_c..deg_b is one contiguous span)
    long n4 = (long)(((char*)hp - (char*)agg_c) / 16);
    k_zero<<<1024, 256, 0, stream>>>((float4*)agg_c, n4);

    // 2) weight prep (transpose to N-major bf16)
    k_build_rel_w<<<(3 * HDIM * FEAT + 255) / 256, 256, 0, stream>>>(basis, coeff, Wt_rel);
    k_transpose_w<<<(HDIM * FEAT + 255) / 256, 256, 0, stream>>>(inp_W_paper, Wt_ip, FEAT, HDIM);
    k_transpose_w<<<(HDIM * 64 + 255) / 256, 256, 0, stream>>>(inp_W_author, Wt_ia, 64, HDIM);

    // 3) input projections (TDM-staged weights + WMMA), bf16 outputs
    k_proj_gemm<<<(N_PAPER / 16 + 7) / 8, 256, 0, stream>>>(
        feat_paper, Wt_ip, inp_b_paper, hp, N_PAPER, FEAT);
    k_proj_gemm<<<(N_AUTHOR / 16 + 7) / 8, 256, 0, stream>>>(
        feat_author, Wt_ia, inp_b_author, ha, N_AUTHOR, 64);

    // 4) edge-wise segment sums (one wave per edge, f32 atomics)
    int sblocks = (int)(((long)NEDGE * 32 + 255) / 256);
    k_scatter<<<sblocks, 256, 0, stream>>>(hp, cites_src,  cites_dst,  agg_c, deg_c, NEDGE);
    k_scatter<<<sblocks, 256, 0, stream>>>(ha, writes_src, writes_dst, agg_w, deg_w, NEDGE);
    k_scatter<<<sblocks, 256, 0, stream>>>(hp, wb_src,     wb_dst,     agg_b, deg_b, NEDGE);

    // 5) fused mean-normalize + relation GEMM(s) + conv_bias + leaky (TDM + WMMA)
    k_rel_gemm<<<(N_PAPER / 16 + 7) / 8, 256, 0, stream>>>(
        agg_c, deg_c, agg_w, deg_w,
        Wt_rel + 0 * HDIM * FEAT, Wt_rel + 1 * HDIM * FEAT,
        conv_bias, agg_c /* = out_p */, N_PAPER, 1);
    k_rel_gemm<<<(N_AUTHOR / 16 + 7) / 8, 256, 0, stream>>>(
        agg_b, deg_b, nullptr, nullptr,
        Wt_rel + 2 * HDIM * FEAT, Wt_rel + 2 * HDIM * FEAT,
        conv_bias, agg_b /* = out_a */, N_AUTHOR, 0);

    // 6) MLP heads + softmax -> d_out (yp then ya, concatenated)
    float* yp = (float*)d_out;
    float* ya = yp + (long)N_PAPER * NCLS;
    k_head<<<(N_PAPER + 255) / 256, 256, 0, stream>>>(agg_c, W1_p, b1_p, W2_p, b2_p, yp, N_PAPER);
    k_head<<<(N_AUTHOR + 255) / 256, 256, 0, stream>>>(agg_b, W1_a, b1_a, W2_a, b2_a, ya, N_AUTHOR);
}
